// DisKT_15152644620332
// MI455X (gfx1250) — matrix operand order repaired and checked by reading, not compiled, not run on hardware
//
#include <hip/hip_runtime.h>
#include <math.h>

// ---------------- model constants ----------------
#define BB   32
#define SS   512
#define DD   512
#define HH   8
#define DHH  64
#define NTOK (BB * SS)            // 16384 rows
#define SCALE_ATTN 0.125f         // 1/sqrt(64)

#define USE_ASYNC_LDS 1           // CDNA5 global_load_async_to_lds path

typedef _Float16 v16h __attribute__((ext_vector_type(16)));
typedef _Float16 v8h  __attribute__((ext_vector_type(8)));
typedef float    v8f  __attribute__((ext_vector_type(8)));

static __device__ inline v16h make_frag(const _Float16* p0, const _Float16* p1) {
    v8h lo = *(const v8h*)p0;
    v8h hi = *(const v8h*)p1;
    v16h r;
#pragma unroll
    for (int i = 0; i < 8; ++i) { r[i] = lo[i]; r[i + 8] = hi[i]; }
    return r;
}

static __device__ inline v8f wmma16(v16h a, v16h b, v8f c) {
    return __builtin_amdgcn_wmma_f32_16x16x32_f16(false, a, false, b, (short)0, c, false, false);
}

#if USE_ASYNC_LDS
// copy 16 bytes global -> LDS through the async unit (ASYNCcnt)
static __device__ inline void async_cp16(const _Float16* g, _Float16* l) {
    unsigned ldsoff = (unsigned)(unsigned long long)l;   // LDS aperture low bits
    asm volatile("global_load_async_to_lds_b128 %0, %1, off"
                 :: "v"(ldsoff), "v"((unsigned long long)g) : "memory");
}
static __device__ inline void async_wait0() {
    asm volatile("s_wait_asynccnt 0x0" ::: "memory");
}
#endif

// =====================================================================
// Generic GEMM: C[M,N] = act( A[M,K] @ W[K,N] + bias (+ C if acc) )
// 64x64x32 tiles, 128 threads (4 waves), LDS double-buffered,
// register-staged fp32 loads -> f16 LDS -> v_wmma_f32_16x16x32_f16.
// act: 0=none, 1=relu, 2=gelu(exact).  outf16: write _Float16 output.
// =====================================================================
__global__ __launch_bounds__(128)
void gemm_wmma_kernel(const float* __restrict__ A, const float* __restrict__ W,
                      const float* __restrict__ bias, void* __restrict__ Cout,
                      int M, int K, int N, int acc, int act, int outf16) {
    __shared__ __align__(16) _Float16 Asub[2][64][40];  // 80B row stride
    __shared__ __align__(16) _Float16 Bsub[2][32][72];  // 144B row stride

    const int tid     = threadIdx.x;
    const int wave    = tid >> 5;
    const int lane    = tid & 31;
    const int rl      = lane & 15;
    const bool lohalf = lane < 16;

    const int rowBase = blockIdx.y * 64;
    const int colBase = blockIdx.x * 64;
    const int ksteps  = K >> 5;

    v8f accf[4];
#pragma unroll
    for (int t = 0; t < 4; ++t) accf[t] = (v8f)0.0f;

    float4 areg[4], breg[4];

    // ---- stage tile kt into registers (all loads issued back-to-back) ----
    auto load_tile = [&](int kt) {
#pragma unroll
        for (int i = 0; i < 4; ++i) {
            int lin = tid + i * 128;          // A: 512 float4 slots
            int r   = lin >> 3;
            int c4  = lin & 7;
            const float* gp = A + (size_t)(rowBase + r) * K + (kt << 5) + (c4 << 2);
            areg[i] = *(const float4*)gp;
            if (kt + 1 < ksteps) __builtin_prefetch(gp + 32, 0, 1);
        }
#pragma unroll
        for (int i = 0; i < 4; ++i) {
            int lin = tid + i * 128;          // B: 512 float4 slots
            int r   = lin >> 4;
            int c4  = lin & 15;
            const float* gp = W + (size_t)((kt << 5) + r) * N + colBase + (c4 << 2);
            breg[i] = *(const float4*)gp;
            if (kt + 1 < ksteps) __builtin_prefetch(gp + (size_t)32 * N, 0, 1);
        }
    };
    // ---- convert + store staged registers into LDS buffer `buf` ----
    auto store_tile = [&](int buf) {
#pragma unroll
        for (int i = 0; i < 4; ++i) {
            int lin = tid + i * 128;
            int r   = lin >> 3;
            int cc  = (lin & 7) << 2;
            Asub[buf][r][cc + 0] = (_Float16)areg[i].x;
            Asub[buf][r][cc + 1] = (_Float16)areg[i].y;
            Asub[buf][r][cc + 2] = (_Float16)areg[i].z;
            Asub[buf][r][cc + 3] = (_Float16)areg[i].w;
        }
#pragma unroll
        for (int i = 0; i < 4; ++i) {
            int lin = tid + i * 128;
            int r   = lin >> 4;
            int cc  = (lin & 15) << 2;
            Bsub[buf][r][cc + 0] = (_Float16)breg[i].x;
            Bsub[buf][r][cc + 1] = (_Float16)breg[i].y;
            Bsub[buf][r][cc + 2] = (_Float16)breg[i].z;
            Bsub[buf][r][cc + 3] = (_Float16)breg[i].w;
        }
    };

    load_tile(0);
    store_tile(0);
    __syncthreads();

    for (int kt = 0; kt < ksteps; ++kt) {
        const int cur = kt & 1;
        const bool more = (kt + 1) < ksteps;
        if (more) load_tile(kt + 1);          // loads in flight during compute

        const int arow = wave * 16 + rl;
        const int kofs = lohalf ? 0 : 8;
        v16h afrag = make_frag(&Asub[cur][arow][kofs], &Asub[cur][arow][kofs + 16]);
#pragma unroll
        for (int nt = 0; nt < 4; ++nt) {
            v16h bfrag;
#pragma unroll
            for (int i = 0; i < 8; ++i) { bfrag[i] = Bsub[cur][kofs + i][nt * 16 + rl]; bfrag[i + 8] = Bsub[cur][kofs + 16 + i][nt * 16 + rl]; }
            accf[nt] = wmma16(afrag, bfrag, accf[nt]);
        }

        if (more) store_tile(cur ^ 1);
        __syncthreads();
    }

    // ---- writeback ----
#pragma unroll
    for (int nt = 0; nt < 4; ++nt) {
#pragma unroll
        for (int v = 0; v < 8; ++v) {
            int r  = rowBase + wave * 16 + (lohalf ? v : v + 8);
            int cx = colBase + nt * 16 + rl;
            size_t idx = (size_t)r * N + cx;
            float val = accf[nt][v];
            if (bias) val += bias[cx];
            if (acc)  val += ((const float*)Cout)[idx];
            if (act == 1)      val = fmaxf(val, 0.0f);
            else if (act == 2) val = 0.5f * val * (1.0f + erff(val * 0.70710678118f));
            if (outf16) ((_Float16*)Cout)[idx] = (_Float16)val;
            else        ((float*)Cout)[idx]    = val;
        }
    }
}

// =====================================================================
// Flash attention with WMMA. One block = (b, h, 64-query tile), 128 thr.
// Q,K,V are f16 in merged [B,S,D] layout; head h -> cols h*64..h*64+63.
// Strict causal (j < i), optional DUAL: key mask cm[b][j]!=0, two V streams.
// Q/V tiles staged through global_load_async_to_lds_b128 (ASYNCcnt);
// K tile is transposed manually (b128 load + b16 scatter).
// =====================================================================
template <bool DUAL>
__global__ __launch_bounds__(128)
void attn_wmma_kernel(const _Float16* __restrict__ Q, const _Float16* __restrict__ Kx,
                      const _Float16* __restrict__ V1, const _Float16* __restrict__ V2,
                      const int* __restrict__ cm,
                      float* __restrict__ O1, float* __restrict__ O2) {
    __shared__ __align__(16) _Float16 Qs [64][72];   // [q][d]
    __shared__ __align__(16) _Float16 Kts[64][72];   // [d][j]  (transposed)
    __shared__ __align__(16) _Float16 Vs [64][72];   // [j][d]
    __shared__ __align__(16) _Float16 Vs2[64][72];   // [j][d]  (dual)
    __shared__ __align__(16) _Float16 Ps [64][72];   // [q][j]
    __shared__ float Ss[64][65];
    __shared__ float mrow[64], lrow[64], arow[64];

    const int tid     = threadIdx.x;
    const int wave    = tid >> 5;
    const int lane    = tid & 31;
    const int rl      = lane & 15;
    const bool lohalf = lane < 16;

    const int qt = blockIdx.x;
    const int h  = blockIdx.y;
    const int b  = blockIdx.z;

    const size_t baseRow = (size_t)b * SS;
    const int    hcol    = h * DHH;

    if (tid < 64) { mrow[tid] = -3.0e38f; lrow[tid] = 0.0f; }

    // ---- stage Q tile: straight 16B-chunk copies ----
#pragma unroll
    for (int i = 0; i < 4; ++i) {
        int lin = tid + i * 128;                     // 512 chunks of 8 halves
        int r   = lin >> 3;
        int d0  = (lin & 7) << 3;
        const _Float16* gp = Q + (baseRow + qt * 64 + r) * DD + hcol + d0;
#if USE_ASYNC_LDS
        async_cp16(gp, &Qs[r][d0]);
#else
        *(v8h*)&Qs[r][d0] = *(const v8h*)gp;
#endif
    }

    v8f o1f[4], o2f[4];
#pragma unroll
    for (int t = 0; t < 4; ++t) { o1f[t] = (v8f)0.0f; o2f[t] = (v8f)0.0f; }

    const int m0 = wave * 16;

    for (int kt = 0; kt <= qt; ++kt) {
        __syncthreads();   // previous-iteration readers done before overwrite
        // ---- V (and V2): async straight copies ----
#pragma unroll
        for (int i = 0; i < 4; ++i) {
            int lin = tid + i * 128;
            int r   = lin >> 3;
            int d0  = (lin & 7) << 3;
            const _Float16* gv = V1 + (baseRow + kt * 64 + r) * DD + hcol + d0;
#if USE_ASYNC_LDS
            async_cp16(gv, &Vs[r][d0]);
#else
            *(v8h*)&Vs[r][d0] = *(const v8h*)gv;
#endif
            if constexpr (DUAL) {
                const _Float16* gv2 = V2 + (baseRow + kt * 64 + r) * DD + hcol + d0;
#if USE_ASYNC_LDS
                async_cp16(gv2, &Vs2[r][d0]);
#else
                *(v8h*)&Vs2[r][d0] = *(const v8h*)gv2;
#endif
            }
        }
        // ---- K: load 8 halves, scatter transposed ----
        {
            v8h kreg[4];
#pragma unroll
            for (int i = 0; i < 4; ++i) {
                int lin = tid + i * 128;
                int r   = lin >> 3;
                int d0  = (lin & 7) << 3;
                kreg[i] = *(const v8h*)(Kx + (baseRow + kt * 64 + r) * DD + hcol + d0);
            }
#pragma unroll
            for (int i = 0; i < 4; ++i) {
                int lin = tid + i * 128;
                int r   = lin >> 3;
                int d0  = (lin & 7) << 3;
#pragma unroll
                for (int e = 0; e < 8; ++e) Kts[d0 + e][r] = kreg[i][e];
            }
        }
#if USE_ASYNC_LDS
        async_wait0();                       // our async copies landed in LDS
#endif
        __syncthreads();

        // ---- S = Q @ K^T  (dh = 64 -> 2 chunks of 32) ----
        {
            const int arow0 = m0 + rl;
            const int kofs  = lohalf ? 0 : 8;
            v16h a0 = make_frag(&Qs[arow0][kofs],      &Qs[arow0][kofs + 16]);
            v16h a1 = make_frag(&Qs[arow0][32 + kofs], &Qs[arow0][32 + kofs + 16]);
#pragma unroll
            for (int nt = 0; nt < 4; ++nt) {
                v8f s = (v8f)0.0f;
                v16h b0, b1;
#pragma unroll
                for (int i = 0; i < 8; ++i) {
                    b0[i] = Kts[kofs + i][nt * 16 + rl];      b0[i + 8] = Kts[kofs + 16 + i][nt * 16 + rl];
                    b1[i] = Kts[32 + kofs + i][nt * 16 + rl]; b1[i + 8] = Kts[32 + kofs + 16 + i][nt * 16 + rl];
                }
                s = wmma16(a0, b0, s);
                s = wmma16(a1, b1, s);
#pragma unroll
                for (int v = 0; v < 8; ++v)
                    Ss[m0 + (lohalf ? v : v + 8)][nt * 16 + rl] = s[v] * SCALE_ATTN;
            }
        }
        __syncthreads();

        // ---- online softmax update (threads 0..63, one row each) ----
        if (tid < 64) {
            const int r  = tid;
            const int qi = qt * 64 + r;
            const int jb = kt * 64;
            float m    = mrow[r];
            float tmax = -3.0e38f;
            for (int j = 0; j < 64; ++j) {
                float s = Ss[r][j];
                bool ok = (jb + j) < qi;
                if constexpr (DUAL) ok = ok && (cm[b * SS + jb + j] != 0);
                if (!ok) s = -3.0e38f;
                Ss[r][j] = s;
                tmax = fmaxf(tmax, s);
            }
            float nm = fmaxf(m, tmax);
            float alpha, suml = 0.0f;
            if (nm <= -1.0e37f) {
                alpha = 1.0f;
                for (int j = 0; j < 64; ++j) Ps[r][j] = (_Float16)0.0f;
            } else {
                alpha = __expf(m - nm);
                for (int j = 0; j < 64; ++j) {
                    float p = __expf(Ss[r][j] - nm);
                    suml += p;
                    Ps[r][j] = (_Float16)p;
                }
            }
            lrow[r] = lrow[r] * alpha + suml;
            mrow[r] = nm;
            arow[r] = alpha;
        }
        __syncthreads();

        // ---- O = O*alpha + P @ V ----
        {
            float al[8];
#pragma unroll
            for (int v = 0; v < 8; ++v) al[v] = arow[m0 + (lohalf ? v : v + 8)];
#pragma unroll
            for (int nt = 0; nt < 4; ++nt)
#pragma unroll
                for (int v = 0; v < 8; ++v) {
                    o1f[nt][v] *= al[v];
                    if constexpr (DUAL) o2f[nt][v] *= al[v];
                }

            const int arow0 = m0 + rl;
            const int kofs  = lohalf ? 0 : 8;
            v16h p0 = make_frag(&Ps[arow0][kofs],      &Ps[arow0][kofs + 16]);
            v16h p1 = make_frag(&Ps[arow0][32 + kofs], &Ps[arow0][32 + kofs + 16]);
#pragma unroll
            for (int nt = 0; nt < 4; ++nt) {
                v16h b0, b1;
#pragma unroll
                for (int i = 0; i < 8; ++i) {
                    b0[i] = Vs[kofs + i][nt * 16 + rl];      b0[i + 8] = Vs[kofs + 16 + i][nt * 16 + rl];
                    b1[i] = Vs[32 + kofs + i][nt * 16 + rl]; b1[i + 8] = Vs[32 + kofs + 16 + i][nt * 16 + rl];
                }
                o1f[nt] = wmma16(p0, b0, o1f[nt]);
                o1f[nt] = wmma16(p1, b1, o1f[nt]);
                if constexpr (DUAL) {
                    v16h c0, c1;
#pragma unroll
                    for (int i = 0; i < 8; ++i) {
                        c0[i] = Vs2[kofs + i][nt * 16 + rl];      c0[i + 8] = Vs2[kofs + 16 + i][nt * 16 + rl];
                        c1[i] = Vs2[32 + kofs + i][nt * 16 + rl]; c1[i + 8] = Vs2[32 + kofs + 16 + i][nt * 16 + rl];
                    }
                    o2f[nt] = wmma16(p0, c0, o2f[nt]);
                    o2f[nt] = wmma16(p1, c1, o2f[nt]);
                }
            }
        }
    }

    // ---- normalize + store (l==0 -> zero row, reproduces row-0 zeroing) ----
#pragma unroll
    for (int nt = 0; nt < 4; ++nt)
#pragma unroll
        for (int v = 0; v < 8; ++v) {
            int r    = m0 + (lohalf ? v : v + 8);
            float lr = lrow[r];
            float inv = (lr > 0.0f) ? (1.0f / lr) : 0.0f;
            size_t g = (baseRow + qt * 64 + r) * DD + hcol + nt * 16 + rl;
            O1[g] = o1f[nt][v] * inv;
            if constexpr (DUAL) O2[g] = o2f[nt][v] * inv;
        }
}

// =====================================================================
// out = LN(a + r)*g + beta ; optional f16 side output
// =====================================================================
__global__ __launch_bounds__(256)
void ln_res_kernel(const float* __restrict__ a, const float* __restrict__ r,
                   const float* __restrict__ g, const float* __restrict__ be,
                   float* __restrict__ out, _Float16* __restrict__ out16, float eps) {
    __shared__ float red[256];
    const int row = blockIdx.x, t = threadIdx.x;
    const size_t base = (size_t)row * DD;
    float v0 = a[base + t] + r[base + t];
    float v1 = a[base + t + 256] + r[base + t + 256];

    red[t] = v0 + v1;
    __syncthreads();
    for (int s = 128; s > 0; s >>= 1) { if (t < s) red[t] += red[t + s]; __syncthreads(); }
    float mu = red[0] * (1.0f / DD);
    __syncthreads();

    float d0 = v0 - mu, d1 = v1 - mu;
    red[t] = d0 * d0 + d1 * d1;
    __syncthreads();
    for (int s = 128; s > 0; s >>= 1) { if (t < s) red[t] += red[t + s]; __syncthreads(); }
    float rinv = rsqrtf(red[0] * (1.0f / DD) + eps);
    __syncthreads();

    float o0 = d0 * rinv * g[t]       + be[t];
    float o1 = d1 * rinv * g[t + 256] + be[t + 256];
    out[base + t]       = o0;
    out[base + t + 256] = o1;
    if (out16) {
        out16[base + t]       = (_Float16)o0;
        out16[base + t + 256] = (_Float16)o1;
    }
}

// =====================================================================
// Rasch embeddings + positional encoding (one block per token)
// =====================================================================
__global__ __launch_bounds__(256)
void rasch_kernel(const float* __restrict__ concept, const float* __restrict__ inter,
                  const float* __restrict__ cvar, const float* __restrict__ qdiffW,
                  const float* __restrict__ ivar,
                  const int* __restrict__ qseq, const int* __restrict__ cseq,
                  const int* __restrict__ rseq,
                  float* __restrict__ x, float* __restrict__ y,
                  float* __restrict__ posi, float* __restrict__ negi,
                  float* __restrict__ qemb, float* __restrict__ qdv) {
    const int tok = blockIdx.x;
    const int c = cseq[tok], q = qseq[tok], r = rseq[tok];
    const int mr = r * ((r > -1) ? 1 : 0);
    const int c2 = mr * c, q2 = mr * q, t2 = 2 - mr;
    const int c3 = (1 - mr) * c, q3 = (1 - mr) * q, t3 = 2 * mr;
    const int s = tok % SS;
    const size_t base = (size_t)tok * DD;
    const float lg = logf(10000.0f) / (float)DD;

#pragma unroll
    for (int i = 0; i < 2; ++i) {
        int d = threadIdx.x + i * 256;
        float ce = concept[(size_t)c * DD + d];
        float qd = qdiffW[(size_t)q * DD + d];
        float qe = ce + qd * cvar[(size_t)c * DD + d];
        float ie = inter[(size_t)mr * DD + d] + ce + qd * ivar[(size_t)mr * DD + d];

        float pi = inter[(size_t)t2 * DD + d] + concept[(size_t)c2 * DD + d]
                 + qdiffW[(size_t)q2 * DD + d] * ivar[(size_t)t2 * DD + d];
        float ni = inter[(size_t)t3 * DD + d] + concept[(size_t)c3 * DD + d]
                 + qdiffW[(size_t)q3 * DD + d] * ivar[(size_t)t3 * DD + d];

        int deven = d & ~1;
        float ang = (float)s * __expf(-(float)deven * lg);
        float pe  = (d & 1) ? __cosf(ang) : __sinf(ang);

        x[base + d]    = qe + pe;
        y[base + d]    = ie + pe;
        posi[base + d] = pi;
        negi[base + d] = ni;
        qemb[base + d] = qe;
        qdv[base + d]  = qd;
    }
}

// final = x - o1 - o2 - qd ; diff = o1 - o2
__global__ __launch_bounds__(256)
void combine_kernel(const float* __restrict__ x, const float* __restrict__ o1,
                    const float* __restrict__ o2, const float* __restrict__ qd,
                    float* __restrict__ fin, float* __restrict__ diff) {
    size_t i = (size_t)blockIdx.x * 256 + threadIdx.x;
    float a = o1[i], b = o2[i];
    fin[i]  = x[i] - a - b - qd[i];
    diff[i] = a - b;
}

// out[row] = sigmoid(dot(h[row], w2) + b2)
__global__ __launch_bounds__(256)
void pred_out_kernel(const float* __restrict__ h, const float* __restrict__ w2,
                     const float* __restrict__ b2, float* __restrict__ out) {
    __shared__ float red[256];
    const int row = blockIdx.x, t = threadIdx.x;
    const size_t base = (size_t)row * DD;
    red[t] = h[base + t] * w2[t] + h[base + t + 256] * w2[t + 256];
    __syncthreads();
    for (int s = 128; s > 0; s >>= 1) { if (t < s) red[t] += red[t + s]; __syncthreads(); }
    if (t == 0) {
        float z = red[0] + b2[0];
        out[row] = 1.0f / (1.0f + __expf(-z));
    }
}

// ---------------------------------------------------------------------
static inline void gemm(const float* A, const float* W, const float* bias,
                        void* C, int M, int K, int N, int acc, int act, int outf16,
                        hipStream_t s) {
    dim3 g(N / 64, M / 64);
    gemm_wmma_kernel<<<g, 128, 0, s>>>(A, W, bias, C, M, K, N, acc, act, outf16);
}

extern "C" void kernel_launch(void* const* d_in, const int* in_sizes, int n_in,
                              void* d_out, int out_size, void* d_ws, size_t ws_size,
                              hipStream_t stream) {
    (void)in_sizes; (void)n_in; (void)out_size; (void)ws_size;
    const float* p_concept = (const float*)d_in[0];
    const float* p_inter   = (const float*)d_in[1];
    const float* p_cvar    = (const float*)d_in[2];
    const float* p_qdiff   = (const float*)d_in[3];
    const float* p_ivar    = (const float*)d_in[4];
    const float* f_w1 = (const float*)d_in[69];
    const float* f_b1 = (const float*)d_in[70];
    const float* f_w2 = (const float*)d_in[71];
    const float* f_b2 = (const float*)d_in[72];
    const float* f_lg = (const float*)d_in[73];
    const float* f_lb = (const float*)d_in[74];
    const float* pr_w1 = (const float*)d_in[75];
    const float* pr_b1 = (const float*)d_in[76];
    const float* pr_w2 = (const float*)d_in[77];
    const float* pr_b2 = (const float*)d_in[78];
    const int* qseq  = (const int*)d_in[79];
    const int* cseq  = (const int*)d_in[80];
    const int* rseq  = (const int*)d_in[81];
    const int* cmseq = (const int*)d_in[82];
    float* out = (float*)d_out;

    // ---- workspace: 13 fp32 slots + 3 f16 slots of [16384 x 512] ----
    const size_t NT = (size_t)NTOK * DD;
    float* ws    = (float*)d_ws;
    float* x     = ws + 0 * NT;
    float* y     = ws + 1 * NT;
    float* posi  = ws + 2 * NT;   // later: final
    float* negi  = ws + 3 * NT;   // later: diff
    float* qemb  = ws + 4 * NT;
    float* qdv   = ws + 5 * NT;
    float* t2    = ws + 8 * NT;   // attn v-scratch is now f16; t2/t3 = o1/o2
    float* t3    = ws + 9 * NT;
    float* t4    = ws + 10 * NT;
    float* hid   = ws + 11 * NT;  // [16384 x 1024] (2 slots)
    _Float16* f0 = (_Float16*)(ws + 13 * NT);  // q  / x(f16)
    _Float16* f1 = f0 + NT;                    // k  / y1(f16)
    _Float16* f2 = f1 + NT;                    // v  / y2(f16)

    rasch_kernel<<<NTOK, 256, 0, stream>>>(p_concept, p_inter, p_cvar, p_qdiff, p_ivar,
                                           qseq, cseq, rseq,
                                           x, y, posi, negi, qemb, qdv);

    dim3 ag(SS / 64, HH, BB);
    for (int nb = 0; nb < 4; ++nb) {
        const int base = 5 + nb * 16;
        const float* Wq = (const float*)d_in[base + 0];
        const float* bq = (const float*)d_in[base + 1];
        const float* Wk = (const float*)d_in[base + 2];
        const float* bk = (const float*)d_in[base + 3];
        const float* Wv = (const float*)d_in[base + 4];
        const float* bv = (const float*)d_in[base + 5];
        const float* Wo = (const float*)d_in[base + 6];
        const float* bo = (const float*)d_in[base + 7];
        const float* l1g = (const float*)d_in[base + 8];
        const float* l1b = (const float*)d_in[base + 9];
        const float* W1 = (const float*)d_in[base + 10];
        const float* b1 = (const float*)d_in[base + 11];
        const float* W2 = (const float*)d_in[base + 12];
        const float* b2 = (const float*)d_in[base + 13];
        const float* l2g = (const float*)d_in[base + 14];
        const float* l2b = (const float*)d_in[base + 15];

        // QKV projections straight to f16 (consumed only by attention)
        gemm(x, Wq, bq, f0, NTOK, DD, DD, 0, 0, 1, stream);
        gemm(x, Wk, bk, f1, NTOK, DD, DD, 0, 0, 1, stream);
        gemm(y, Wv, bv, f2, NTOK, DD, DD, 0, 0, 1, stream);

        attn_wmma_kernel<false><<<ag, 128, 0, stream>>>(f0, f1, f2, nullptr, nullptr,
                                                        t3, nullptr);

        gemm(t3, Wo, bo, t4, NTOK, DD, DD, 0, 0, 0, stream);
        ln_res_kernel<<<NTOK, 256, 0, stream>>>(x, t4, l1g, l1b, x, nullptr, 1e-5f);

        gemm(x, W1, b1, hid, NTOK, DD, 2 * DD, 0, /*relu*/1, 0, stream);
        gemm(hid, W2, b2, t4, NTOK, 2 * DD, DD, 0, 0, 0, stream);
        // last block: also emit x in f16 for the dual attention (q = k = x)
        ln_res_kernel<<<NTOK, 256, 0, stream>>>(x, t4, l2g, l2b, x,
                                                (nb == 3) ? f0 : nullptr, 1e-5f);
    }

    // shared FFN streams: y1 = FFN(posi) -> f1 (f16), y2 = FFN(negi) -> f2 (f16)
    gemm(posi, f_w1, f_b1, hid, NTOK, DD, 2 * DD, 0, /*gelu*/2, 0, stream);
    gemm(hid, f_w2, f_b2, t4, NTOK, 2 * DD, DD, 0, 0, 0, stream);
    ln_res_kernel<<<NTOK, 256, 0, stream>>>(t4, posi, f_lg, f_lb, t2, f1, 1e-12f);

    gemm(negi, f_w1, f_b1, hid, NTOK, DD, 2 * DD, 0, /*gelu*/2, 0, stream);
    gemm(hid, f_w2, f_b2, t4, NTOK, 2 * DD, DD, 0, 0, 0, stream);
    ln_res_kernel<<<NTOK, 256, 0, stream>>>(t4, negi, f_lg, f_lb, t2, f2, 1e-12f);

    // dual attention: shared weights, counter mask, two value streams
    attn_wmma_kernel<true><<<ag, 128, 0, stream>>>(f0, f0, f1, f2, cmseq, t2, t3);

    // combine: final = x - o1 - o2 - qdiff (-> posi) ; diff = o1 - o2 (-> negi)
    combine_kernel<<<(unsigned)(NT / 256), 256, 0, stream>>>(x, t2, t3, qdv, posi, negi);

    // prediction head: h = relu([final, q_emb, diff] @ W1 + b1) via 3 acc-GEMMs
    const float* W1a = pr_w1;
    const float* W1b = pr_w1 + (size_t)512 * DD;
    const float* W1c = pr_w1 + (size_t)1024 * DD;
    gemm(posi, W1a, nullptr, t4, NTOK, DD, DD, 0, 0, 0, stream);
    gemm(qemb, W1b, nullptr, t4, NTOK, DD, DD, 1, 0, 0, stream);
    gemm(negi, W1c, pr_b1,  t4, NTOK, DD, DD, 1, /*relu*/1, 0, stream);

    pred_out_kernel<<<NTOK, 256, 0, stream>>>(t4, pr_w2, pr_b2, out);
}
